// DiffusionModule_37529424233116
// MI455X (gfx1250) — matrix operand order repaired and checked by reading, not compiled, not run on hardware
//
#include <hip/hip_runtime.h>
#include <math.h>

// ---------------------------------------------------------------------------
// DiffusionModule for MI455X (gfx1250, wave32, WMMA).
//   N=50000 nodes, E=600000 edges, HID=128, IN_DIM=128, TIME_EMB=32.
//   GEMMs:  v_wmma_f32_16x16x32_bf16 (bf16 in, fp32 accumulate), LDS-resident
//           weights, scheduler-pinned [16x ds_read -> 8x wmma] pipeline.
//   Graph aggregation: float4 gather + global_atomic_add_f32 scatter
//   (bandwidth-bound part: ~6.7 GB total @ 23.3 TB/s).
// Workspace layout (floats): G[N*128] Agg[N*128] H1[N*128] H2[N*128]
//                            dinv1[N] dinv2[N] coef[2] g1row[128]
// Requires ws_size >= (4*N*128 + 2*N + 130) * 4 bytes (~103 MB).
// ---------------------------------------------------------------------------

#define HID 128

typedef __attribute__((ext_vector_type(16))) __bf16 v16bf;
typedef __attribute__((ext_vector_type(8)))  float  v8f;

#if defined(__has_builtin)
#if __has_builtin(__builtin_amdgcn_cvt_pk_bf16_f32)
#define HAVE_CVT_PK_BF16 1
#endif
#if __has_builtin(__builtin_amdgcn_sched_group_barrier)
#define HAVE_SCHED_GROUP 1
#endif
#endif

// Pack two f32 into one dword of bf16 (lo in [15:0], hi in [31:16]).
__device__ __forceinline__ unsigned pack2bf(float lo, float hi) {
#ifdef HAVE_CVT_PK_BF16
  auto p = __builtin_amdgcn_cvt_pk_bf16_f32(lo, hi);
  return __builtin_bit_cast(unsigned, p);
#else
  unsigned ul = __builtin_bit_cast(unsigned, lo) + 0x8000u;  // round-to-nearest
  unsigned uh = __builtin_bit_cast(unsigned, hi) + 0x8000u;
  return (ul >> 16) | (uh & 0xFFFF0000u);
#endif
}

// ---------------------------------------------------------------------------
// C[N x 128] = A[N x 128] @ W[128 x 128]  (+ C if accumulate) (+ add_row bcst)
// 256 threads = 8 waves; one 16-row tile per wave, full 128 output columns.
// ---------------------------------------------------------------------------
__global__ __launch_bounds__(256)
void gemm128_wmma_kernel(const float* __restrict__ A,
                         const float* __restrict__ W,
                         float* __restrict__ C,
                         int n_row_tiles,
                         const float* __restrict__ add_row,
                         int accumulate) {
  // W staged to LDS as bf16, column-major: wlds[col*128 + k].
  // Packed b32 stores: dword i = column i>>6, K pair (i&63)*2.
  __shared__ __attribute__((aligned(32))) __bf16 wlds[HID * HID];
  int tid = threadIdx.x;
  unsigned* wl32 = (unsigned*)wlds;
  for (int i = tid; i < HID * HID / 2; i += 256) {
    int c = i >> 6, kp = (i & 63) << 1;
    wl32[i] = pack2bf(W[(size_t)kp * HID + c], W[(size_t)(kp + 1) * HID + c]);
  }
  __syncthreads();

  int wave = tid >> 5;
  int lane = tid & 31;
  int row_tile = blockIdx.x * 8 + wave;
  if (row_tile >= n_row_tiles) return;          // wave-uniform: EXEC stays full

  int row0 = row_tile * 16;
  int m    = lane & 15;                         // row within tile / column lane
  int half = lane >> 4;
  const float* arow = A + (size_t)(row0 + m) * HID;

  v8f acc[8];
  if (accumulate) {
#pragma unroll
    for (int ct = 0; ct < 8; ++ct)
#pragma unroll
      for (int r = 0; r < 8; ++r)
        acc[ct][r] = C[(size_t)(row0 + r + half * 8) * HID + ct * 16 + m];
  } else {
#pragma unroll
    for (int ct = 0; ct < 8; ++ct) acc[ct] = (v8f){0, 0, 0, 0, 0, 0, 0, 0};
  }

#pragma unroll
  for (int ks = 0; ks < 4; ++ks) {              // K = 128 in steps of 32
    int k0 = ks * 32;
    // A fragment: contiguous 16-elem K run per lane half (k = k0 + 16*half + s).
    int kA = k0 + half * 16;
    int kB = kA + 8;
    float4 a0 = *(const float4*)(arow + kA);
    float4 a1 = *(const float4*)(arow + kA + 4);
    float4 b0 = *(const float4*)(arow + kB);
    float4 b1 = *(const float4*)(arow + kB + 4);
    union { v16bf v; unsigned u[8]; } af;
    af.u[0] = pack2bf(a0.x, a0.y);
    af.u[1] = pack2bf(a0.z, a0.w);
    af.u[2] = pack2bf(a1.x, a1.y);
    af.u[3] = pack2bf(a1.z, a1.w);
    af.u[4] = pack2bf(b0.x, b0.y);
    af.u[5] = pack2bf(b0.z, b0.w);
    af.u[6] = pack2bf(b1.x, b1.y);
    af.u[7] = pack2bf(b1.z, b1.w);

    // Load all 8 B fragments into distinct registers, then run 8 WMMAs.
    v16bf bfr[8];
#pragma unroll
    for (int ct = 0; ct < 8; ++ct)
      bfr[ct] = *(const v16bf*)&wlds[(ct * 16 + m) * HID + k0 + half * 16];
#pragma unroll
    for (int ct = 0; ct < 8; ++ct)
      acc[ct] = __builtin_amdgcn_wmma_f32_16x16x32_bf16(
          false, af.v, false, bfr[ct], (short)0, acc[ct], false, false);

#ifdef HAVE_SCHED_GROUP
    // Pin the schedule: 16 DS reads (0x100) ahead of 8 WMMAs (0x8) so the
    // loads keep distinct registers and waits become partial dscnt waits,
    // overlapping LDS fetch with the matrix pipe. Other instruction classes
    // (A-fragment VALU packing, VMEM) remain free to co-schedule.
    __builtin_amdgcn_sched_group_barrier(0x100, 16, 0);
    __builtin_amdgcn_sched_group_barrier(0x008, 8, 0);
#endif
  }

#pragma unroll
  for (int ct = 0; ct < 8; ++ct) {
    int col = ct * 16 + m;
    float ra = add_row ? add_row[col] : 0.0f;
#pragma unroll
    for (int r = 0; r < 8; ++r)
      C[(size_t)(row0 + r + half * 8) * HID + col] = acc[ct][r] + ra;
  }
}

// ---------------------------------------------------------------------------
// Degree / normalization
// ---------------------------------------------------------------------------
__global__ void init_deg_kernel(float* deg, int n) {
  int i = blockIdx.x * 256 + threadIdx.x;
  if (i < n) deg[i] = 1.0f;                      // self-loop contributes 1
}
__global__ void count_deg_kernel(float* deg, const int* dst, int e) {
  int i = blockIdx.x * 256 + threadIdx.x;
  if (i < e) atomicAdd(&deg[dst[i]], 1.0f);
}
__global__ void fin_deg_kernel(float* dinv, int n) {
  int i = blockIdx.x * 256 + threadIdx.x;
  if (i < n) dinv[i] = rsqrtf(dinv[i]);          // deg >= 1 always
}

// ---------------------------------------------------------------------------
// Edge aggregation: Y[dst] += H[src] * dinv[src]*dinv[dst]   (Y pre-zeroed)
// 32 threads per edge, float4 per thread.
// ---------------------------------------------------------------------------
__global__ void agg_edges_kernel(const float* __restrict__ H,
                                 const int* __restrict__ src,
                                 const int* __restrict__ dst,
                                 const float* __restrict__ dinv,
                                 float* __restrict__ Y, int e) {
  int tid = blockIdx.x * 256 + threadIdx.x;
  int ei = tid >> 5;
  if (ei >= e) return;
  int c0 = (tid & 31) << 2;
  int s = src[ei], d = dst[ei];
  float w = dinv[s] * dinv[d];
  float4 h = *(const float4*)(H + (size_t)s * HID + c0);
  float* y = Y + (size_t)d * HID + c0;
  atomicAdd(y + 0, h.x * w);
  atomicAdd(y + 1, h.y * w);
  atomicAdd(y + 2, h.z * w);
  atomicAdd(y + 3, h.w * w);
}

// Self-loop + bias + optional ReLU:  Y = act(Y + H*dinv^2 + b)
__global__ void finalize_kernel(float* __restrict__ Y,
                                const float* __restrict__ H,
                                const float* __restrict__ dinv,
                                const float* __restrict__ b,
                                int n, int relu) {
  int idx = blockIdx.x * 256 + threadIdx.x;
  if (idx >= n * HID) return;
  int i = idx >> 7, c = idx & 127;
  float di = dinv[i];
  float v = Y[idx] + H[idx] * di * di + b[c];
  Y[idx] = relu ? fmaxf(v, 0.0f) : v;
}

// ---------------------------------------------------------------------------
// Beta-schedule coefficients + time-embedding row + (t_emb @ g1_Wc) row.
// Single block of 128 threads.
// ---------------------------------------------------------------------------
__global__ void prep_kernel(const int* __restrict__ tptr,
                            const float* __restrict__ tmW1,
                            const float* __restrict__ tmb1,
                            const float* __restrict__ tmW2,
                            const float* __restrict__ tmb2,
                            const float* __restrict__ g1Wc,
                            float* __restrict__ coef,
                            float* __restrict__ g1row) {
  __shared__ float e32[32];
  __shared__ float emb[128];
  int t = *tptr;
  int tid = threadIdx.x;
  if (tid == 0) {
    float ab = 1.0f, sab = 0.0f, s1m = 0.0f;
    for (int i = 0; i < 100; ++i) {
      float beta = 1e-4f + (0.02f - 1e-4f) * (float)i / 99.0f;
      ab *= (1.0f - beta);
      if (i == t) { sab = sqrtf(ab); s1m = sqrtf(1.0f - ab); }
    }
    coef[0] = sab;
    coef[1] = s1m;
  }
  float tf = (float)t;
  if (tid < 32) {
    float x = tf * tmW1[tid] + tmb1[tid];
    e32[tid] = x / (1.0f + expf(-x));            // silu
  }
  __syncthreads();
  float te = tmb2[tid];
  for (int j = 0; j < 32; ++j) te += e32[j] * tmW2[j * HID + tid];
  emb[tid] = te;
  __syncthreads();
  float r = 0.0f;
  for (int j = 0; j < 128; ++j) r += emb[j] * g1Wc[j * HID + tid];
  g1row[tid] = r;                                // constant row: t_emb @ Wc
}

// h_t = sab*h0 + s1m*noise  (in place)
__global__ void mix_kernel(float* __restrict__ H, const float* __restrict__ Z,
                           const float* __restrict__ coef, int n) {
  int i = blockIdx.x * 256 + threadIdx.x;
  if (i < n) H[i] = coef[0] * H[i] + coef[1] * Z[i];
}

__global__ void hdiff_kernel(const float* __restrict__ A,
                             const float* __restrict__ B,
                             float* __restrict__ out, int n) {
  int i = blockIdx.x * 256 + threadIdx.x;
  if (i < n) out[i] = 0.5f * (A[i] + B[i]);
}

__global__ void loss_kernel(const float* __restrict__ P,
                            const float* __restrict__ Z,
                            float* __restrict__ out, float scale, int n) {
  __shared__ float red[256];
  int i = blockIdx.x * 256 + threadIdx.x;
  float v = 0.0f;
  if (i < n) { float d = P[i] - Z[i]; v = d * d * scale; }
  red[threadIdx.x] = v;
  __syncthreads();
  for (int s = 128; s > 0; s >>= 1) {
    if (threadIdx.x < s) red[threadIdx.x] += red[threadIdx.x + s];
    __syncthreads();
  }
  if (threadIdx.x == 0) atomicAdd(out, red[0]);
}

// ---------------------------------------------------------------------------
extern "C" void kernel_launch(void* const* d_in, const int* in_sizes, int n_in,
                              void* d_out, int out_size, void* d_ws, size_t ws_size,
                              hipStream_t stream) {
  const float* X      = (const float*)d_in[0];
  const int*   ei1    = (const int*)d_in[1];
  const int*   ei2    = (const int*)d_in[2];
  const int*   tptr   = (const int*)d_in[3];
  const float* noise1 = (const float*)d_in[4];
  const float* noise2 = (const float*)d_in[5];
  const float* encW1  = (const float*)d_in[6];
  const float* encb1  = (const float*)d_in[7];
  const float* encW2  = (const float*)d_in[8];
  const float* encb2  = (const float*)d_in[9];
  const float* tmW1   = (const float*)d_in[10];
  const float* tmb1   = (const float*)d_in[11];
  const float* tmW2   = (const float*)d_in[12];
  const float* tmb2   = (const float*)d_in[13];
  const float* g1W    = (const float*)d_in[14];
  const float* g1b    = (const float*)d_in[15];
  const float* g2W    = (const float*)d_in[16];
  const float* g2b    = (const float*)d_in[17];
  const float* g3W    = (const float*)d_in[18];
  const float* g3b    = (const float*)d_in[19];

  const int N = in_sizes[0] / HID;     // 50000 (multiple of 16)
  const int E = in_sizes[1] / 2;       // 600000
  const size_t NB = (size_t)N * HID;

  float* ws    = (float*)d_ws;
  float* G     = ws;                   // GEMM output (pre-aggregation)
  float* Agg   = G + NB;               // aggregation output / pred
  float* H1    = Agg + NB;             // h0_v1 -> h_t_v1
  float* H2    = H1 + NB;              // h0_v2 -> h_t_v2
  float* dinv1 = H2 + NB;
  float* dinv2 = dinv1 + N;
  float* coef  = dinv2 + N;            // [sqrt_ab, sqrt_1m]
  float* g1row = coef + 2;             // t_emb @ g1_Wc (128)

  float* out    = (float*)d_out;
  float* lossp  = out + NB;

  const int tiles   = N / 16;                    // 3125 full 16-row tiles
  const int gemm_bk = (tiles + 7) / 8;
  const int nb_N    = (N + 255) / 256;
  const int nb_NB   = (int)((NB + 255) / 256);
  const int nb_E    = (E + 255) / 256;
  const int nb_E32  = (int)(((size_t)E * 32 + 255) / 256);

  auto gemm = [&](const float* Ain, const float* Win, float* Cout,
                  const float* addrow, int acc) {
    gemm128_wmma_kernel<<<gemm_bk, 256, 0, stream>>>(Ain, Win, Cout, tiles, addrow, acc);
  };
  auto gcn_agg = [&](const float* Hin, float* Yout, const int* ei,
                     const float* dinv, const float* b, int relu) {
    hipMemsetAsync(Yout, 0, NB * sizeof(float), stream);
    agg_edges_kernel<<<nb_E32, 256, 0, stream>>>(Hin, ei, ei + E, dinv, Yout, E);
    finalize_kernel<<<nb_NB, 256, 0, stream>>>(Yout, Hin, dinv, b, N, relu);
  };

  // --- GCN normalization (symmetric, with self-loops) ---
  init_deg_kernel<<<nb_N, 256, 0, stream>>>(dinv1, N);
  init_deg_kernel<<<nb_N, 256, 0, stream>>>(dinv2, N);
  count_deg_kernel<<<nb_E, 256, 0, stream>>>(dinv1, ei1 + E, E);
  count_deg_kernel<<<nb_E, 256, 0, stream>>>(dinv2, ei2 + E, E);
  fin_deg_kernel<<<nb_N, 256, 0, stream>>>(dinv1, N);
  fin_deg_kernel<<<nb_N, 256, 0, stream>>>(dinv2, N);

  // --- coefficients + time embedding row (t_emb @ g1_Wc) ---
  prep_kernel<<<1, 128, 0, stream>>>(tptr, tmW1, tmb1, tmW2, tmb2,
                                     g1W + 256 * HID, coef, g1row);

  // --- encoders: h0 = GCN2(relu(GCN1(X))) ---
  gemm(X, encW1, G, nullptr, 0);
  gcn_agg(G, Agg, ei1, dinv1, encb1, 1);
  gemm(Agg, encW2, G, nullptr, 0);
  gcn_agg(G, H1, ei1, dinv1, encb2, 0);

  gemm(X, encW1, G, nullptr, 0);
  gcn_agg(G, Agg, ei2, dinv2, encb1, 1);
  gemm(Agg, encW2, G, nullptr, 0);
  gcn_agg(G, H2, ei2, dinv2, encb2, 0);

  // --- h_diff output, then reuse H1/H2 for the noised states ---
  hdiff_kernel<<<nb_NB, 256, 0, stream>>>(H1, H2, out, (int)NB);
  mix_kernel<<<nb_NB, 256, 0, stream>>>(H1, noise1, coef, (int)NB);
  mix_kernel<<<nb_NB, 256, 0, stream>>>(H2, noise2, coef, (int)NB);

  hipMemsetAsync(lossp, 0, sizeof(float), stream);
  const float scale = 1.0f / (float)NB;

  // --- denoiser view 1: concat GEMM folded into two K=128 GEMMs + const row
  gemm(H1, g1W, G, nullptr, 0);
  gemm(H2, g1W + 128 * HID, G, g1row, 1);
  gcn_agg(G, Agg, ei1, dinv1, g1b, 1);
  gemm(Agg, g2W, G, nullptr, 0);
  gcn_agg(G, Agg, ei1, dinv1, g2b, 1);
  gemm(Agg, g3W, G, nullptr, 0);
  gcn_agg(G, Agg, ei1, dinv1, g3b, 0);           // pred_v1
  loss_kernel<<<nb_NB, 256, 0, stream>>>(Agg, noise1, lossp, scale, (int)NB);

  // --- denoiser view 2 (roles swapped) ---
  gemm(H2, g1W, G, nullptr, 0);
  gemm(H1, g1W + 128 * HID, G, g1row, 1);
  gcn_agg(G, Agg, ei2, dinv2, g1b, 1);
  gemm(Agg, g2W, G, nullptr, 0);
  gcn_agg(G, Agg, ei2, dinv2, g2b, 1);
  gemm(Agg, g3W, G, nullptr, 0);
  gcn_agg(G, Agg, ei2, dinv2, g3b, 0);           // pred_v2
  loss_kernel<<<nb_NB, 256, 0, stream>>>(Agg, noise2, lossp, scale, (int)NB);
}
